// GatedEncoderLayer_22634477650317
// MI455X (gfx1250) — matrix operand order repaired and checked by reading, not compiled, not run on hardware
//
#include <hip/hip_runtime.h>

typedef __attribute__((ext_vector_type(16))) __bf16 v16bf;
typedef __attribute__((ext_vector_type(8)))  __bf16 v8bf;
typedef __attribute__((ext_vector_type(8)))  float  v8f;
typedef __attribute__((ext_vector_type(4)))  float  v4f;

namespace {
constexpr int kBatch = 16, kI = 2048, kJ = 1024, kK = 1024, kF = 32;
constexpr int kRows = kBatch * kI;          // 32768 independent output rows
constexpr int kWaves = 8;                   // waves per block (wave32)
constexpr int kThreads = kWaves * 32;       // 256 threads
constexpr int kRowsPerBlock = kWaves * 16;  // one 16-row WMMA tile per wave
constexpr int kJChunk = 512;                // Wy^T LDS staging chunk (J dim)
constexpr int kWyStride = kJChunk + 8;      // 520 bf16: bank stride 260 dw == 4 mod 64
constexpr int kTStride = 34;                // f32 row stride for T staging (conflict-free)
}

union BFrag { v16bf v; v8bf h[2]; };

__global__ __launch_bounds__(kThreads)
void gated_encoder_fused_bf16wmma(const float* __restrict__ X,
                                  const float* __restrict__ Wx,
                                  const float* __restrict__ Wy,
                                  const float* __restrict__ Wz,
                                  float* __restrict__ Out)
{
  __shared__ __bf16 sWyT[kF * kWyStride];          // 33,280 B: Wy^T bf16 chunk
  __shared__ float  sT[kWaves * 16 * kTStride];    // 17,408 B: per-wave T transpose tile

  const int tid  = threadIdx.x;
  const int wave = tid >> 5;
  const int lane = tid & 31;
  const int lo   = lane & 15;   // N column / M row within fragment
  const int hi   = lane >> 4;   // half-wave select

  const int rowBase = blockIdx.x * kRowsPerBlock + wave * 16;
  const float* xrow = X + (size_t)(rowBase + lo) * kJ;

  const int c0 = hi * 8;    // A-fragment K sub-offset (lanes 0-15: 0, 16-31: 8)
  const int bk = hi * 8;    // B-fragment K sub-offset (lanes 0-15: 0, 16-31: 8)

  v8f acc0 = {0.f,0.f,0.f,0.f,0.f,0.f,0.f,0.f};   // t[:, 0:16]
  v8f acc1 = acc0;                                 // t[:, 16:32]

  // ---------------- Stage 1: t[16,32] = X_tile[16,1024] * Wy[1024,32] ----------------
  for (int jb = 0; jb < kJ; jb += kJChunk) {
    __syncthreads();   // previous chunk fully consumed
    for (int idx = tid; idx < kJChunk * kF; idx += kThreads) {
      const int jj = idx >> 5, f = idx & 31;       // global coalesced read of Wy
      sWyT[f * kWyStride + jj] = (__bf16)Wy[jb * kF + idx];
    }
    __syncthreads();

    #pragma unroll 4
    for (int kb = 0; kb < kJChunk; kb += 32) {
      const float* xp = xrow + jb + kb + c0;
      __builtin_prefetch(xp + 512, 0, 1);          // global_prefetch_b8, stream X ahead
      // A fragment (16-bit A 16x32 layout): per lane two 8-elem runs of this row
      const v4f xa0 = *(const v4f*)(xp);
      const v4f xa1 = *(const v4f*)(xp + 4);
      const v4f xa2 = *(const v4f*)(xp + 16);
      const v4f xa3 = *(const v4f*)(xp + 20);
      v16bf a;
      #pragma unroll
      for (int e = 0; e < 4; ++e) {
        a[e]      = (__bf16)xa0[e];
        a[4 + e]  = (__bf16)xa1[e];
        a[8 + e]  = (__bf16)xa2[e];
        a[12 + e] = (__bf16)xa3[e];
      }
      // B fragments: lane holds column f = (half*16 + lo), two 8-elem K runs (bk, bk+16)
      BFrag b0, b1;
      const __bf16* w0 = &sWyT[lo * kWyStride + kb + bk];
      const __bf16* w1 = &sWyT[(16 + lo) * kWyStride + kb + bk];
      b0.h[0] = *(const v8bf*)w0;  b0.h[1] = *(const v8bf*)(w0 + 16);
      b1.h[0] = *(const v8bf*)w1;  b1.h[1] = *(const v8bf*)(w1 + 16);

      acc0 = __builtin_amdgcn_wmma_f32_16x16x32_bf16(false, a, false, b0.v,
                                                     (short)0, acc0, false, false);
      acc1 = __builtin_amdgcn_wmma_f32_16x16x32_bf16(false, a, false, b1.v,
                                                     (short)0, acc1, false, false);
    }
  }

  // ---------------- Gate: t *= Wx (C/D layout: N = lo, M = v + 8*hi) ----------------
  const float* wxp = Wx + (size_t)rowBase * kF;
  #pragma unroll
  for (int v = 0; v < 8; ++v) {
    const int m = v + 8 * hi;
    acc0[v] *= wxp[m * kF + lo];
    acc1[v] *= wxp[m * kF + 16 + lo];
  }

  // ------------- Transpose t from C-layout to A-layout via per-wave LDS tile -------------
  float* tb = &sT[wave * 16 * kTStride];
  #pragma unroll
  for (int v = 0; v < 8; ++v) {
    const int m = v + 8 * hi;
    tb[m * kTStride + lo]      = acc0[v];
    tb[m * kTStride + 16 + lo] = acc1[v];
  }
  __syncthreads();

  const float* trow = tb + lo * kTStride;   // A-layout: lane row M = lo
  v16bf a2;
  #pragma unroll
  for (int e = 0; e < 8; ++e) {
    a2[e]     = (__bf16)trow[c0 + e];        // K = c0 + e
    a2[8 + e] = (__bf16)trow[c0 + 16 + e];   // K = c0 + 16 + e
  }

  // ---------------- Stage 2: out[16,1024] = t[16,32] * Wz^T (K = F = 32) ----------------
  for (int nt = 0; nt < kK / 16; ++nt) {
    // B fragment: lane column = nt*16 + lo, two 8-elem f runs (bk, bk+16)
    const float* wzr = Wz + (size_t)(nt * 16 + lo) * kF + bk;
    const v4f w0 = *(const v4f*)(wzr);
    const v4f w1 = *(const v4f*)(wzr + 4);
    const v4f w2 = *(const v4f*)(wzr + 16);
    const v4f w3 = *(const v4f*)(wzr + 20);
    v16bf b2;
    #pragma unroll
    for (int e = 0; e < 4; ++e) {
      b2[e]      = (__bf16)w0[e];
      b2[4 + e]  = (__bf16)w1[e];
      b2[8 + e]  = (__bf16)w2[e];
      b2[12 + e] = (__bf16)w3[e];
    }
    v8f d = {0.f,0.f,0.f,0.f,0.f,0.f,0.f,0.f};
    d = __builtin_amdgcn_wmma_f32_16x16x32_bf16(false, a2, false, b2,
                                                (short)0, d, false, false);
    // D layout: lane col = nt*16 + lo, VGPR v -> row v + 8*hi (64B contiguous per row half)
    float* orow = Out + (size_t)rowBase * kK + nt * 16 + lo;
    #pragma unroll
    for (int v = 0; v < 8; ++v) {
      orow[(size_t)(v + 8 * hi) * kK] = d[v];
    }
  }
}

extern "C" void kernel_launch(void* const* d_in, const int* in_sizes, int n_in,
                              void* d_out, int out_size, void* d_ws, size_t ws_size,
                              hipStream_t stream) {
  const float* X  = (const float*)d_in[0];   // (16, 2048, 1024) f32
  const float* Wx = (const float*)d_in[1];   // (16, 2048, 32) f32
  const float* Wy = (const float*)d_in[2];   // (1024, 32) f32
  const float* Wz = (const float*)d_in[3];   // (1024, 32) f32
  float* Out = (float*)d_out;                // (16, 2048, 1024) f32

  dim3 grid(kRows / kRowsPerBlock);          // 256 blocks
  dim3 block(kThreads);                      // 256 threads = 8 waves
  gated_encoder_fused_bf16wmma<<<grid, block, 0, stream>>>(X, Wx, Wy, Wz, Out);
}
